// KernelPointAggregation_39694087749727
// MI455X (gfx1250) — compile-verified, matrix operand order checked by reading, not run on hardware
//
#include <hip/hip_runtime.h>
#include <hip/hip_bf16.h>

typedef __attribute__((ext_vector_type(16))) _Float16 v16h;
typedef __attribute__((ext_vector_type(8)))  float    v8f;

#define MINV 1e-12f
#define EPSV 1e-6f
#define MAXN 0.99999f        /* (1 - 1e-5) / SQC, C = SQC = 1 */
#define KPE  0.66f

// ---------------- workspace layout ----------------
// half region (fragment-major f16 weights), then float region.
#define HW1   0            // 64x256   -> 16384 halves
#define HFW1  16384        // 64x128   ->  8192
#define HFW2  24576        // 128x128  -> 16384
#define HGW1  40960        // 128x64   ->  8192
#define HGW2  49152        // 64x64    ->  4096
#define HALF_COUNT 53248
#define HALF_BYTES (HALF_COUNT * 2)   // 106496, 256B aligned

// float region offsets (in floats, from float base)
#define OF_KLP     0        // 4x64
#define OF_KLPN2   256      // 4
#define OF_HB      260      // 4x64
#define OF_HBN2    516      // 4
#define OF_EB1     520      // 128
#define OF_EB1N2   648      // 1
#define OF_EB2     649      // 128
#define OF_EB2N2   777      // 1
#define OF_EGB1    778      // 64
#define OF_EGB1N2  842      // 1
#define OF_EGB2    843      // 64
#define OF_EGB2N2  907      // 1
#define OF_MIDN2   908      // 16000
#define OF_MID     16912    // 16000x128

// ---------------- helpers ----------------
static __device__ inline float wred(float v) {
  #pragma unroll
  for (int off = 16; off >= 1; off >>= 1) v += __shfl_xor(v, off, 32);
  return v;
}

static __device__ inline float artanhc(float v) {
  v = fminf(fmaxf(v, -1.f + EPSV), 1.f - EPSV);
  return 0.5f * logf((1.f + v) / (1.f - v));
}

// A fragment: 16x32 f16 slice of a row-major f32 LDS matrix (16 rows).
// lane L<16 holds row M=L, K = {k0..k0+7, k0+16..k0+23}; lane L+16: K += 8.
static __device__ inline v16h load_a(const float* A, int astr, int k0, int lane) {
  int m  = lane & 15;
  int kb = k0 + ((lane & 16) >> 1);
  const float* r = A + m * astr + kb;
  v16h a;
  #pragma unroll
  for (int j = 0; j < 8; ++j) {
    a[j]     = (_Float16)r[j];
    a[j + 8] = (_Float16)r[16 + j];
  }
  return a;
}

// B fragment: weights pre-packed fragment-major -> one contiguous 32B load.
static __device__ inline v16h load_b(const _Float16* W, int Nc, int ks, int nt, int lane) {
  const _Float16* p = W + ((((size_t)ks * (Nc >> 4) + nt) * 32 + lane) << 4);
  return *(const v16h*)p;
}

// D = A(16 x kdim, f32 LDS) * W(kdim x Nc, packed f16) -> O (16 x Nc, f32 LDS)
static __device__ inline void mm(const float* A, int astr, int kdim,
                                 const _Float16* W, int Nc,
                                 float* O, int ostr, int lane) {
  int mb = (lane & 16) ? 8 : 0;
  int nl = lane & 15;
  int nts = Nc >> 4, kss = kdim >> 5;
  for (int nt = 0; nt < nts; ++nt) {
    v8f acc = {};
    for (int ks = 0; ks < kss; ++ks) {
      v16h a = load_a(A, astr, ks * 32, lane);
      v16h b = load_b(W, Nc, ks, nt, lane);
      acc = __builtin_amdgcn_wmma_f32_16x16x32_f16(false, a, false, b,
                                                   (short)0, acc, false, false);
    }
    #pragma unroll
    for (int j = 0; j < 8; ++j) O[(j + mb) * ostr + nt * 16 + nl] = acc[j];
  }
}

// Row-wise Mobius-linear tail for 16 rows in LDS (after mx = x @ W.T):
// res = proj(mobius_from_mx(mx, x)); res = proj(mobius_add(res, hb));
// optionally res = proj(expmap0(relu(logmap0(res)))).
template <int EPL>
static __device__ void rowwise_blinear(float* buf, int stride,
                                       const float* inn2, float* outn2,
                                       const float* biasv, float bn2,
                                       bool do_act, int lane) {
  for (int m = 0; m < 16; ++m) {
    float v[EPL], bv[EPL];
    #pragma unroll
    for (int e = 0; e < EPL; ++e) {
      v[e]  = buf[m * stride + e * 32 + lane];
      bv[e] = biasv[e * 32 + lane];
    }
    float xn = sqrtf(fmaxf(inn2[m], MINV));
    float sq = 0.f;
    #pragma unroll
    for (int e = 0; e < EPL; ++e) sq += v[e] * v[e];
    sq = wred(sq);
    float mxn = sqrtf(fmaxf(sq, MINV));
    float t = tanhf(mxn / xn * artanhc(xn));
    float s = t / mxn;
    float p1 = (t > MAXN) ? MAXN / t : 1.f;
    s *= p1;
    float f2 = t * p1; f2 *= f2;
    float dp = 0.f;
    #pragma unroll
    for (int e = 0; e < EPL; ++e) dp += v[e] * bv[e];
    dp = wred(dp);
    float xy = s * dp;
    float aa = 1.f + 2.f * xy + bn2, bb = 1.f - f2;
    float den = fmaxf(1.f + 2.f * xy + f2 * bn2, MINV);
    float n2 = (aa * aa * f2 + 2.f * aa * bb * xy + bb * bb * bn2) / (den * den);
    float nr = sqrtf(fmaxf(n2, MINV));
    float p2 = (nr > MAXN) ? MAXN / nr : 1.f;
    float o[EPL];
    #pragma unroll
    for (int e = 0; e < EPL; ++e) o[e] = p2 * (aa * s * v[e] + bb * bv[e]) / den;
    float r2 = n2 * p2 * p2;
    if (do_act) {
      float rn = sqrtf(fmaxf(r2, MINV));
      float sl = artanhc(rn) / rn;
      float rq = 0.f;
      #pragma unroll
      for (int e = 0; e < EPL; ++e) { o[e] = fmaxf(0.f, sl * o[e]); rq += o[e] * o[e]; }
      rq = wred(rq);
      float rn2 = sqrtf(fmaxf(rq, MINV));
      float tt = tanhf(rn2);
      float se = tt / rn2;
      float p3 = (tt > MAXN) ? MAXN / tt : 1.f;
      #pragma unroll
      for (int e = 0; e < EPL; ++e) o[e] *= se * p3;
      r2 = tt * p3; r2 *= r2;
    }
    #pragma unroll
    for (int e = 0; e < EPL; ++e) buf[m * stride + e * 32 + lane] = o[e];
    if (lane == 0) outn2[m] = r2;
  }
}

// ---------------- prep kernels ----------------
static __device__ inline void fragpos(int idx, int Nc, int& ki, int& c) {
  int j = idx & 15;
  int q = idx >> 4;
  int lane = q & 31; q >>= 5;
  int nt = q % (Nc >> 4), kb = q / (Nc >> 4);
  ki = kb * 32 + ((lane & 16) >> 1) + (j < 8 ? j : j + 8);
  c  = nt * 16 + (lane & 15);
}

__global__ void kp_prep_pack(const float* __restrict__ lin_w,
                             const float* __restrict__ fw1,
                             const float* __restrict__ fw2,
                             const float* __restrict__ gw1,
                             const float* __restrict__ gw2,
                             _Float16* __restrict__ H) {
  int idx = blockIdx.x * blockDim.x + threadIdx.x;
  if (idx >= HALF_COUNT) return;
  int ki, c;
  float v;
  if (idx < HFW1) {                    // W1[i][k*64+o] = lin_w[k][o][i]
    fragpos(idx, 256, ki, c);
    v = lin_w[(c >> 6) * 4096 + (c & 63) * 64 + ki];
  } else if (idx < HFW2) {             // FW1[i][j] = fw1[j][i]
    fragpos(idx - HFW1, 128, ki, c);
    v = fw1[c * 64 + ki];
  } else if (idx < HGW1) {             // FW2[i][j] = fw2[j][i]
    fragpos(idx - HFW2, 128, ki, c);
    v = fw2[c * 128 + ki];
  } else if (idx < HGW2) {             // GW1[i][j] = gw1[j][i]
    fragpos(idx - HGW1, 64, ki, c);
    v = gw1[c * 128 + ki];
  } else {                             // GW2[i][j] = gw2[j][i]
    fragpos(idx - HGW2, 64, ki, c);
    v = gw2[c * 64 + ki];
  }
  H[idx] = (_Float16)v;
}

static __device__ inline void emap_proj(const float* v, int dim, int i,
                                        float* dst, float* n2out) {
  float s = 0.f;
  for (int j = 0; j < dim; ++j) s += v[j] * v[j];
  float n  = sqrtf(fmaxf(s, MINV));
  float th = tanhf(n);
  float sc = th / n;
  float nrm = sc * sqrtf(s);
  float p = (nrm > MAXN) ? MAXN / nrm : 1.f;
  dst[i] = p * sc * v[i];
  if (i == 0) n2out[0] = (nrm * p) * (nrm * p);
}

__global__ void kp_prep_vec(const float* __restrict__ kt,
                            const float* __restrict__ lin_b,
                            const float* __restrict__ fb1,
                            const float* __restrict__ fb2,
                            const float* __restrict__ gb1,
                            const float* __restrict__ gb2,
                            float* __restrict__ F) {
  for (int t = threadIdx.x; t < 896; t += blockDim.x) {
    if (t < 256) {                      // klp = expmap0(kt)  (no proj)
      int k = t >> 6, i = t & 63;
      const float* v = kt + k * 64;
      float s = 0.f;
      for (int j = 0; j < 64; ++j) s += v[j] * v[j];
      float n  = sqrtf(fmaxf(s, MINV));
      float th = tanhf(n);
      F[OF_KLP + t] = th / n * v[i];
      if (i == 0) { float nrm = th / n * sqrtf(s); F[OF_KLPN2 + k] = nrm * nrm; }
    } else if (t < 512) {               // hb = proj(expmap0(lin_b))
      int u = t - 256; int k = u >> 6, i = u & 63;
      emap_proj(lin_b + k * 64, 64, i, F + OF_HB + k * 64, F + OF_HBN2 + k);
    } else if (t < 640) {
      emap_proj(fb1, 128, t - 512, F + OF_EB1, F + OF_EB1N2);
    } else if (t < 768) {
      emap_proj(fb2, 128, t - 640, F + OF_EB2, F + OF_EB2N2);
    } else if (t < 832) {
      emap_proj(gb1, 64, t - 768, F + OF_EGB1, F + OF_EGB1N2);
    } else {
      emap_proj(gb2, 64, t - 832, F + OF_EGB2, F + OF_EGB2N2);
    }
  }
}

// ---------------- main per-node kernel: one wave32 per node ----------------
__global__ void __launch_bounds__(32)
kp_nodes(const float* __restrict__ x, const int* __restrict__ nei,
         const float* __restrict__ nmask, const _Float16* __restrict__ H,
         float* __restrict__ F) {
  const _Float16* W1h  = H + HW1;
  const _Float16* FW1h = H + HFW1;
  const _Float16* FW2h = H + HFW2;
  const float* klp   = F + OF_KLP;
  const float* klpn2 = F + OF_KLPN2;
  const float* hb    = F + OF_HB;
  const float* hbn2  = F + OF_HBN2;
  const float* eb1   = F + OF_EB1;
  const float* eb2   = F + OF_EB2;
  float eb1n2 = F[OF_EB1N2], eb2n2 = F[OF_EB2N2];

  __shared__ float s_xr[64];
  __shared__ float s_xnei[16][68];
  __shared__ float s_x0[16][68];
  __shared__ float s_mx[16][260];
  __shared__ float s_agg[16][68];
  __shared__ float s_h[16][132];
  __shared__ float s_h2[16][132];
  __shared__ float s_w[4][16];
  __shared__ float s_rn2[16], s_x0n2[16], s_aggn2[16], s_hn2[16], s_h2n2[16];

  int n = blockIdx.x, lane = threadIdx.x;

  // node point: proj(x[n])
  float a0 = x[n * 64 + lane], a1 = x[n * 64 + 32 + lane];
  float s2 = wred(a0 * a0 + a1 * a1);
  float nn = sqrtf(fmaxf(s2, MINV));
  float ps = (nn > MAXN) ? MAXN / nn : 1.f;
  a0 *= ps; a1 *= ps;
  s_xr[lane] = a0; s_xr[32 + lane] = a1;
  float xr2 = s2 * ps * ps;

  // gather neighbors, proj, transport to origin frame: x0 = proj(mobius_add(-xr, xn))
  for (int m = 0; m < 16; ++m) {
    int idx = nei[n * 16 + m];
    float b0 = x[idx * 64 + lane], b1 = x[idx * 64 + 32 + lane];
    float t2 = wred(b0 * b0 + b1 * b1);
    float tn = sqrtf(fmaxf(t2, MINV));
    float tps = (tn > MAXN) ? MAXN / tn : 1.f;
    b0 *= tps; b1 *= tps;
    s_xnei[m][lane] = b0; s_xnei[m][32 + lane] = b1;
    float y2 = t2 * tps * tps;
    float xy = wred(-(a0 * b0 + a1 * b1));            // dot(-xr, y)
    float aa = 1.f + 2.f * xy + y2;
    float bb = 1.f - xr2;
    float den = fmaxf(1.f + 2.f * xy + xr2 * y2, MINV);
    float n2 = (aa * aa * xr2 + 2.f * aa * bb * xy + bb * bb * y2) / (den * den);
    float nr = sqrtf(fmaxf(n2, MINV));
    float pp = (nr > MAXN) ? MAXN / nr : 1.f;
    float sc = pp / den;
    s_x0[m][lane]      = sc * (aa * (-a0) + bb * b0);
    s_x0[m][32 + lane] = sc * (aa * (-a1) + bb * b1);
    if (lane == 0) { s_rn2[m] = y2; s_x0n2[m] = n2 * pp * pp; }
  }
  __syncthreads();

  // kernel-point correlation weights: d = 2*artanh(|mobius_add(-x0, klp)|)
  for (int m = 0; m < 16; ++m) {
    float xv0 = s_x0[m][lane], xv1 = s_x0[m][32 + lane];
    float xx2 = s_x0n2[m];
    float wk[4]; float wsum = 0.f;
    #pragma unroll
    for (int k = 0; k < 4; ++k) {
      float xy = wred(-(xv0 * klp[k * 64 + lane] + xv1 * klp[k * 64 + 32 + lane]));
      float y2 = klpn2[k];
      float aa = 1.f + 2.f * xy + y2, bb = 1.f - xx2;
      float den = fmaxf(1.f + 2.f * xy + xx2 * y2, MINV);
      float n2 = (aa * aa * xx2 + 2.f * aa * bb * xy + bb * bb * y2) / (den * den);
      float nr = sqrtf(fmaxf(n2, MINV));
      float d = 2.f * artanhc(nr);
      float w = fmaxf(0.f, 1.f - d * (1.f / KPE));
      wk[k] = w; wsum += w;
    }
    float inv = 1.f / fmaxf(wsum, MINV);
    if (lane == 0) {
      s_w[0][m] = wk[0] * inv; s_w[1][m] = wk[1] * inv;
      s_w[2][m] = wk[2] * inv; s_w[3][m] = wk[3] * inv;
    }
  }
  __syncthreads();

  // mx[n,k,m,o] : 16x64 @ 64x256 via WMMA
  mm(&s_xnei[0][0], 68, 64, W1h, 256, &s_mx[0][0], 260, lane);
  __syncthreads();

  // per-kernel mobius linear tail + Klein midpoint over kernels -> agg
  for (int m = 0; m < 16; ++m) {
    float xn = sqrtf(fmaxf(s_rn2[m], MINV));
    float atx = artanhc(xn);
    float e0 = 0.f, e1 = 0.f, dsum = 0.f;
    #pragma unroll
    for (int k = 0; k < 4; ++k) {
      float m0 = s_mx[m][k * 64 + lane], m1 = s_mx[m][k * 64 + 32 + lane];
      float mq = wred(m0 * m0 + m1 * m1);
      float mxn = sqrtf(fmaxf(mq, MINV));
      float t = tanhf(mxn / xn * atx);
      float s = t / mxn;
      float p1 = (t > MAXN) ? MAXN / t : 1.f;
      s *= p1;
      float f2 = t * p1; f2 *= f2;
      float hb0 = hb[k * 64 + lane], hb1 = hb[k * 64 + 32 + lane];
      float xy = s * wred(m0 * hb0 + m1 * hb1);
      float y2 = hbn2[k];
      float aa = 1.f + 2.f * xy + y2, bb = 1.f - f2;
      float den = fmaxf(1.f + 2.f * xy + f2 * y2, MINV);
      float n2 = (aa * aa * f2 + 2.f * aa * bb * xy + bb * bb * y2) / (den * den);
      float nr = sqrtf(fmaxf(n2, MINV));
      float p2 = (nr > MAXN) ? MAXN / nr : 1.f;
      float o0 = p2 * (aa * s * m0 + bb * hb0) / den;
      float o1 = p2 * (aa * s * m1 + bb * hb1) / den;
      float ff2 = n2 * p2 * p2;
      float kscale = 2.f / (1.f + ff2);
      float kn2 = kscale * kscale * ff2;
      float gam = 1.f / sqrtf(fmaxf(1.f - kn2, MINV));
      float wg = s_w[k][m] * gam;
      e0 += wg * kscale * o0; e1 += wg * kscale * o1; dsum += wg;
    }
    float inv = 1.f / fmaxf(dsum, MINV);
    e0 *= inv; e1 *= inv;
    float kn2 = wred(e0 * e0 + e1 * e1);
    float dk = 1.f + sqrtf(fmaxf(1.f - kn2, MINV));
    float p0 = e0 / dk, p1v = e1 / dk;
    float pn2 = kn2 / (dk * dk);
    float pn = sqrtf(fmaxf(pn2, MINV));
    float pp = (pn > MAXN) ? MAXN / pn : 1.f;
    s_agg[m][lane] = p0 * pp; s_agg[m][32 + lane] = p1v * pp;
    if (lane == 0) s_aggn2[m] = pn2 * pp * pp;
  }
  __syncthreads();

  // blinear(agg, fw1, fb1, relu) -> h [16x128]
  mm(&s_agg[0][0], 68, 64, FW1h, 128, &s_h[0][0], 132, lane);
  __syncthreads();
  rowwise_blinear<4>(&s_h[0][0], 132, s_aggn2, s_hn2, eb1, eb1n2, true, lane);
  __syncthreads();

  // blinear(h, fw2, fb2) -> h2 [16x128]
  mm(&s_h[0][0], 132, 128, FW2h, 128, &s_h2[0][0], 132, lane);
  __syncthreads();
  rowwise_blinear<4>(&s_h2[0][0], 132, s_hn2, s_h2n2, eb2, eb2n2, false, lane);
  __syncthreads();

  // Klein midpoint over neighbors -> mid[n] (128)
  float ea[4] = {0.f, 0.f, 0.f, 0.f};
  float dsum = 0.f;
  for (int m = 0; m < 16; ++m) {
    float wm = nmask[n * 16 + m];
    float f2 = s_h2n2[m];
    float kscale = 2.f / (1.f + f2);
    float kn2 = kscale * kscale * f2;
    float gam = 1.f / sqrtf(fmaxf(1.f - kn2, MINV));
    float wg = wm * gam;
    #pragma unroll
    for (int e = 0; e < 4; ++e) ea[e] += wg * kscale * s_h2[m][e * 32 + lane];
    dsum += wg;
  }
  float inv = 1.f / fmaxf(dsum, MINV);
  float kq = 0.f;
  #pragma unroll
  for (int e = 0; e < 4; ++e) { ea[e] *= inv; kq += ea[e] * ea[e]; }
  kq = wred(kq);
  float dk = 1.f + sqrtf(fmaxf(1.f - kq, MINV));
  float pn2 = kq / (dk * dk);
  float pn = sqrtf(fmaxf(pn2, MINV));
  float pp = (pn > MAXN) ? MAXN / pn : 1.f;
  float* mid = F + OF_MID + (size_t)n * 128;
  #pragma unroll
  for (int e = 0; e < 4; ++e) mid[e * 32 + lane] = ea[e] * (pp / dk);
  if (lane == 0) F[OF_MIDN2 + n] = pn2 * pp * pp;
}

// ---------------- final kernel: 16 nodes per wave ----------------
__global__ void __launch_bounds__(32)
kp_final(const float* __restrict__ F, const _Float16* __restrict__ H,
         float* __restrict__ out) {
  const _Float16* GW1h = H + HGW1;
  const _Float16* GW2h = H + HGW2;
  const float* egb1 = F + OF_EGB1;
  const float* egb2 = F + OF_EGB2;
  float b1n2 = F[OF_EGB1N2], b2n2 = F[OF_EGB2N2];

  __shared__ float s_mid[16][132];
  __shared__ float s_t[16][68];
  __shared__ float s_o[16][68];
  __shared__ float s_mn2[16], s_tn2[16], s_on2[16];

  int lane = threadIdx.x;
  int nb = blockIdx.x * 16;

  for (int r = 0; r < 16; ++r) {
    const float* src = F + OF_MID + (size_t)(nb + r) * 128;
    #pragma unroll
    for (int e = 0; e < 4; ++e) s_mid[r][e * 32 + lane] = src[e * 32 + lane];
  }
  if (lane < 16) s_mn2[lane] = F[OF_MIDN2 + nb + lane];
  __syncthreads();

  // blinear(mid, gw1, gb1, relu) -> 16x64
  mm(&s_mid[0][0], 132, 128, GW1h, 64, &s_t[0][0], 68, lane);
  __syncthreads();
  rowwise_blinear<2>(&s_t[0][0], 68, s_mn2, s_tn2, egb1, b1n2, true, lane);
  __syncthreads();

  // blinear(t, gw2, gb2) -> 16x64
  mm(&s_t[0][0], 68, 64, GW2h, 64, &s_o[0][0], 68, lane);
  __syncthreads();
  rowwise_blinear<2>(&s_o[0][0], 68, s_tn2, s_on2, egb2, b2n2, false, lane);
  __syncthreads();

  for (int r = 0; r < 16; ++r) {
    out[(size_t)(nb + r) * 64 + lane]      = s_o[r][lane];
    out[(size_t)(nb + r) * 64 + 32 + lane] = s_o[r][32 + lane];
  }
}

// ---------------- launch ----------------
extern "C" void kernel_launch(void* const* d_in, const int* in_sizes, int n_in,
                              void* d_out, int out_size, void* d_ws, size_t ws_size,
                              hipStream_t stream) {
  (void)in_sizes; (void)n_in; (void)out_size; (void)ws_size;
  const float* x        = (const float*)d_in[0];
  const int*   nei      = (const int*)d_in[1];
  const float* nei_mask = (const float*)d_in[2];
  const float* kt       = (const float*)d_in[3];
  const float* lin_w    = (const float*)d_in[4];
  const float* lin_b    = (const float*)d_in[5];
  const float* fw1      = (const float*)d_in[6];
  const float* fb1      = (const float*)d_in[7];
  const float* fw2      = (const float*)d_in[8];
  const float* fb2      = (const float*)d_in[9];
  const float* gw1      = (const float*)d_in[10];
  const float* gb1      = (const float*)d_in[11];
  const float* gw2      = (const float*)d_in[12];
  const float* gb2      = (const float*)d_in[13];

  _Float16* H = (_Float16*)d_ws;
  float*    F = (float*)((char*)d_ws + HALF_BYTES);
  float*  out = (float*)d_out;

  kp_prep_pack<<<HALF_COUNT / 256, 256, 0, stream>>>(lin_w, fw1, fw2, gw1, gw2, H);
  kp_prep_vec<<<1, 256, 0, stream>>>(kt, lin_b, fb1, fb2, gb1, gb2, F);
  kp_nodes<<<16000, 32, 0, stream>>>(x, nei, nei_mask, H, F);
  kp_final<<<1000, 32, 0, stream>>>(F, H, out);
}
